// LongformerAttention_10170482557216
// MI455X (gfx1250) — compile-verified, matrix operand order checked by reading, not run on hardware
//
#include <hip/hip_runtime.h>
#include <hip/hip_bf16.h>

// ---------------------------------------------------------------------------
// Types for CDNA5 WMMA (wave32)
// ---------------------------------------------------------------------------
typedef __bf16 bf16_t;
typedef __attribute__((ext_vector_type(16))) __bf16 v16bf;
typedef __attribute__((ext_vector_type(8)))  __bf16 v8bf;
typedef __attribute__((ext_vector_type(8)))  float  v8f;
typedef __attribute__((ext_vector_type(4)))  unsigned int v4u;
typedef __attribute__((ext_vector_type(8)))  int    v8i;
typedef __attribute__((ext_vector_type(4)))  int    v4i;

#define B_    2
#define S_    4096
#define E_    1024
#define H_    16
#define D_    64
#define W_    256
#define NC_   16

// Tensor Data Mover availability (device pass only; host pass parses fallback)
#if defined(__HIP_DEVICE_COMPILE__) && __has_builtin(__builtin_amdgcn_tensor_load_to_lds)
#define HAVE_TDM 1
#else
#define HAVE_TDM 0
#endif

// ---------------------------------------------------------------------------
// Fragment helpers.
//
// 16-bit A-matrix 16x32 per-lane layout (ISA 7.12.2): lane L handles row
// r = L&15; halves 0..7 hold K = kh..kh+7, halves 8..15 hold K = 16+kh..16+kh+7
// with kh = (L>>4)*8.  B 32x16 mirrors this with lane = column N.
// Loading a row-major 16x32 bf16 tile therefore needs two contiguous 16-byte
// loads per lane for either operand.
// ---------------------------------------------------------------------------
__device__ __forceinline__ v16bf load_frag(const bf16_t* base, int ld, int lane) {
    const int r  = lane & 15;
    const int kh = (lane >> 4) << 3;
    const v8bf a = *(const v8bf*)(base + (size_t)r * ld + kh);
    const v8bf b = *(const v8bf*)(base + (size_t)r * ld + kh + 16);
    return __builtin_shufflevector(a, b, 0,1,2,3,4,5,6,7,8,9,10,11,12,13,14,15);
}

__device__ __forceinline__ v8f wmma_bf16(v16bf a, v16bf b, v8f c) {
    // D = A(16x32) * B(32x16) + C, fp32 accumulate
    return __builtin_amdgcn_wmma_f32_16x16x32_bf16(false, a, false, b,
                                                   (short)0, c, false, false);
}

__device__ __forceinline__ v8f zero8() {
    v8f z;
#pragma unroll
    for (int i = 0; i < 8; ++i) z[i] = 0.0f;
    return z;
}

#if HAVE_TDM
// ---------------------------------------------------------------------------
// TDM: DMA one 128-row x 32-col bf16 tile (row stride = ld elements) from
// global memory into LDS.  D# per ISA ch.8:
//   group0: [1:0]=count=1, [63:32]=lds byte addr, [120:64]=global addr,
//           [127:126]=type=2
//   group1: [17:16]=data_size=1 (2 bytes), tensor_dim0/1 large (no OOB clip),
//           [127:112]=tile_dim0=32, [143:128]=tile_dim1=128,
//           [207:160]=tensor_dim0_stride=ld
// Issued by one wave per WG; completion via s_wait_tensorcnt + barrier.
// ---------------------------------------------------------------------------
__device__ __forceinline__ void tdm_load_tile(unsigned int lds_off,
                                              const bf16_t* gsrc,
                                              unsigned int ld_elems) {
    const unsigned long long ga = (unsigned long long)(uintptr_t)gsrc;
    v4u g0;
    g0[0] = 1u;                                   // count=1 (valid descriptor)
    g0[1] = lds_off;                              // lds_addr (bytes)
    g0[2] = (unsigned int)ga;                     // global_addr[31:0]
    g0[3] = (unsigned int)((ga >> 32) & 0x01FFFFFFu) | 0x80000000u; // addr[56:32] | type=2
    const unsigned int td0 = 1u << 20, td1 = 1u << 20;  // generous extents
    v8i g1;
    g1[0] = 0x00010000;                                   // data_size=2B
    g1[1] = (int)((td0 & 0xFFFFu) << 16);                 // tensor_dim0 lo16
    g1[2] = (int)((td0 >> 16) | ((td1 & 0xFFFFu) << 16)); // tdim0 hi / tdim1 lo
    g1[3] = (int)((td1 >> 16) | (32u << 16));             // tdim1 hi / tile_dim0=32
    g1[4] = 128;                                          // tile_dim1=128, tile_dim2=0
    g1[5] = (int)ld_elems;                                // tensor_dim0_stride lo32
    g1[6] = 0;                                            // stride hi16 / dim1_stride lo
    g1[7] = 0;
    v4i g2; g2[0] = 0; g2[1] = 0; g2[2] = 0; g2[3] = 0;
    v4i g3; g3[0] = 0; g3[1] = 0; g3[2] = 0; g3[3] = 0;
#if __clang_major__ >= 23
    v8i gz; 
#pragma unroll
    for (int i = 0; i < 8; ++i) gz[i] = 0;
    __builtin_amdgcn_tensor_load_to_lds(g0, g1, g2, g3, gz, 0);
#else
    __builtin_amdgcn_tensor_load_to_lds(g0, g1, g2, g3, 0);
#endif
}
#endif  // HAVE_TDM

// ---------------------------------------------------------------------------
// fp32 -> bf16 convert
// ---------------------------------------------------------------------------
__global__ void cvt_f32_bf16(const float* __restrict__ in, bf16_t* __restrict__ out, int n) {
    int i = blockIdx.x * blockDim.x + threadIdx.x;
    if (i < n) out[i] = (bf16_t)in[i];
}

// ---------------------------------------------------------------------------
// GEMM: out(MxN) = A(MxK) * W(NxK)^T + bias
// MODE 0: store bf16 row-major (M x N)
// MODE 1: store bf16 "transposed V" layout: vt[(b*E + n)*S + spos], m=b*S+spos
// MODE 2: store fp32 row-major
// Block = 128x128 tile, 8 waves; wave = 32x64 (2 M-tiles x 4 N-tiles).
// A/W 128x32 slabs staged through LDS by the Tensor Data Mover
// (double-buffered, wave 0 drives the DMA) when available.
// ---------------------------------------------------------------------------
template <int MODE>
__global__ void __launch_bounds__(256)
gemm_wmma(const bf16_t* __restrict__ A, const bf16_t* __restrict__ Wt,
          const float* __restrict__ bias, void* __restrict__ out,
          int M, int N, int K) {
    const int lane = threadIdx.x & 31;
    const int wv   = threadIdx.x >> 5;     // 0..7
    const int mblocks = M >> 7;
    const int mb = blockIdx.x % mblocks;
    const int nb = blockIdx.x / mblocks;
    const int m0 = (mb << 7) + ((wv >> 1) << 5);   // wave's first row  (32 rows)
    const int n0 = (nb << 7) + ((wv &  1) << 6);   // wave's first col  (64 cols)

    v8f acc[2][4];
#pragma unroll
    for (int t = 0; t < 2; ++t)
#pragma unroll
        for (int u = 0; u < 4; ++u) acc[t][u] = zero8();

#if HAVE_TDM
    __shared__ bf16_t sA[2][128 * 32];
    __shared__ bf16_t sB[2][128 * 32];
    const bf16_t* gA = A  + (size_t)(mb << 7) * K;
    const bf16_t* gB = Wt + (size_t)(nb << 7) * K;
    if (wv == 0) {
        tdm_load_tile((unsigned int)(uintptr_t)sA[0], gA, (unsigned int)K);
        tdm_load_tile((unsigned int)(uintptr_t)sB[0], gB, (unsigned int)K);
    }
    int buf = 0;
    for (int k = 0; k < K; k += 32, buf ^= 1) {
        if (wv == 0) {
            if (k + 32 < K) {
                // prefetch next slab into the other buffer, then wait for the
                // current slab (TDM completes in order per wave: 2 outstanding
                // descriptors belong to the prefetch)
                tdm_load_tile((unsigned int)(uintptr_t)sA[buf ^ 1], gA + k + 32, (unsigned int)K);
                tdm_load_tile((unsigned int)(uintptr_t)sB[buf ^ 1], gB + k + 32, (unsigned int)K);
                __builtin_amdgcn_s_wait_tensorcnt(2);
            } else {
                __builtin_amdgcn_s_wait_tensorcnt(0);
            }
        }
        __syncthreads();                    // current slab visible to all waves
        const bf16_t* a0 = sA[buf] + (size_t)((wv >> 1) << 5) * 32;
        const bf16_t* b0 = sB[buf] + (size_t)((wv &  1) << 6) * 32;
        v16bf af[2], bfr[4];
#pragma unroll
        for (int t = 0; t < 2; ++t)
            af[t] = load_frag(a0 + (size_t)(16 * t) * 32, 32, lane);
#pragma unroll
        for (int u = 0; u < 4; ++u)
            bfr[u] = load_frag(b0 + (size_t)(16 * u) * 32, 32, lane);
#pragma unroll
        for (int t = 0; t < 2; ++t)
#pragma unroll
            for (int u = 0; u < 4; ++u)
                acc[t][u] = wmma_bf16(af[t], bfr[u], acc[t][u]);
        __syncthreads();                    // all reads done before overwrite
    }
#else
    for (int k = 0; k < K; k += 32) {
        if (k + 64 < K) {
            __builtin_prefetch(A  + (size_t)m0 * K + k + 64, 0, 1);
            __builtin_prefetch(Wt + (size_t)n0 * K + k + 64, 0, 1);
        }
        v16bf af[2], bfr[4];
#pragma unroll
        for (int t = 0; t < 2; ++t)
            af[t] = load_frag(A + (size_t)(m0 + 16 * t) * K + k, K, lane);
#pragma unroll
        for (int u = 0; u < 4; ++u)
            bfr[u] = load_frag(Wt + (size_t)(n0 + 16 * u) * K + k, K, lane);
#pragma unroll
        for (int t = 0; t < 2; ++t)
#pragma unroll
            for (int u = 0; u < 4; ++u)
                acc[t][u] = wmma_bf16(af[t], bfr[u], acc[t][u]);
    }
#endif

    // Epilogue: C fragment lane L holds column n0+16u+(L&15); VGPR i holds row
    // m0+16t+(L>>4)*8+i.
#pragma unroll
    for (int u = 0; u < 4; ++u) {
        const int n = n0 + 16 * u + (lane & 15);
        const float bn = bias[n];
#pragma unroll
        for (int t = 0; t < 2; ++t) {
            const int row0 = m0 + 16 * t + ((lane >> 4) << 3);
#pragma unroll
            for (int i = 0; i < 8; ++i) {
                const float val = acc[t][u][i] + bn;
                const int m = row0 + i;
                if (MODE == 0) {
                    ((bf16_t*)out)[(size_t)m * N + n] = (bf16_t)val;
                } else if (MODE == 1) {
                    const int bb = m >> 12;          // m / S_
                    const int sp = m & (S_ - 1);     // m % S_
                    ((bf16_t*)out)[((size_t)bb * E_ + n) * S_ + sp] = (bf16_t)val;
                } else {
                    ((float*)out)[(size_t)m * N + n] = val;
                }
            }
        }
    }
}

// ---------------------------------------------------------------------------
// In-place RoPE over the head axis (reference rotates by head index, not
// position): ang(i) = head * 10000^(-(i mod 32)/32), interleaved pairing.
// Applies `scale` after rotation (q: 1/sqrt(D), k: 1).
// ---------------------------------------------------------------------------
__global__ void rope_kernel(bf16_t* __restrict__ t, float scale) {
    const int total = B_ * S_ * H_ * (D_ / 2);
    int idx = blockIdx.x * blockDim.x + threadIdx.x;
    if (idx >= total) return;
    const int p   = idx & 31;            // pair within head (0..31)
    const int hh  = (idx >> 5) & 15;     // head
    const int tok = idx >> 9;            // b*S + s
    const size_t base = (size_t)tok * E_ + hh * D_;
    const int i0 = 2 * p, i1 = 2 * p + 1;
    const float x0 = (float)t[base + i0];
    const float x1 = (float)t[base + i1];
    const float inv0 = powf(10000.0f, -((float)(i0 & 31)) / 32.0f);
    const float inv1 = powf(10000.0f, -((float)(i1 & 31)) / 32.0f);
    const float a0 = (float)hh * inv0;
    const float a1 = (float)hh * inv1;
    const float o0 = x0 * __cosf(a0) - x1 * __sinf(a0);
    const float o1 = x1 * __cosf(a1) + x0 * __sinf(a1);
    t[base + i0] = (bf16_t)(o0 * scale);
    t[base + i1] = (bf16_t)(o1 * scale);
}

// ---------------------------------------------------------------------------
// Sliding-window attention, one WG per (b, h, chunk); 8 waves x 32 queries.
// Scores computed transposed (S^T = K * Q^T) so the f32 C-fragment lines up
// with the bf16 A-fragment packing of the P*V matmul without any LDS bounce.
// Online (flash) softmax in fp32.
// ---------------------------------------------------------------------------
__global__ void __launch_bounds__(256)
attn_kernel(const bf16_t* __restrict__ q, const bf16_t* __restrict__ k,
            const bf16_t* __restrict__ vt, bf16_t* __restrict__ ctx) {
    const int lane = threadIdx.x & 31;
    const int wv   = threadIdx.x >> 5;
    const int c = blockIdx.x & 15;
    const int h = (blockIdx.x >> 4) & 15;
    const int b = blockIdx.x >> 8;
    const int qbase = wv * 32;

    // Q fragments (B-operand of S^T): lane = query row, contiguous dims.
    v16bf qf[2][2];
#pragma unroll
    for (int qt = 0; qt < 2; ++qt)
#pragma unroll
        for (int ks = 0; ks < 2; ++ks)
            qf[qt][ks] = load_frag(
                q + (size_t)(b * S_ + c * W_ + qbase + qt * 16) * E_ + h * D_ + ks * 32,
                E_, lane);

    v8f acc[2][4];
#pragma unroll
    for (int qt = 0; qt < 2; ++qt)
#pragma unroll
        for (int nt = 0; nt < 4; ++nt) acc[qt][nt] = zero8();
    float mrun[2] = {-1e30f, -1e30f};
    float lrun[2] = {0.0f, 0.0f};

    // Window is 3 chunks [c-1, c, c+1]; out-of-sequence chunks are
    // block-aligned -> just shrink the y-range (no padding needed).
    const int y0 = (c == 0)        ? W_     : 0;
    const int y1 = (c == NC_ - 1)  ? 2 * W_ : 3 * W_;
    const int qx0 = qbase + (lane & 15);     // in-chunk query idx (per lane)
    const int kh8 = (lane >> 4) << 3;

    for (int yb = y0; yb < y1; yb += 32) {
        const int kp = c * W_ + yb - W_;     // global key position of block
        // K A-fragments: 2 key-groups x 2 k-steps over D
        v16bf ka[2][2];
#pragma unroll
        for (int g = 0; g < 2; ++g)
#pragma unroll
            for (int ks = 0; ks < 2; ++ks)
                ka[g][ks] = load_frag(
                    k + (size_t)(b * S_ + kp + g * 16) * E_ + h * D_ + ks * 32,
                    E_, lane);
        // V B-fragments from transposed V (dim-major): lane = dim col,
        // halves = 32 contiguous keys.
        v16bf vb[4];
#pragma unroll
        for (int nt = 0; nt < 4; ++nt)
            vb[nt] = load_frag(
                vt + ((size_t)(b * H_ + h) * D_ + nt * 16) * S_ + kp,
                S_, lane);

#pragma unroll
        for (int qt = 0; qt < 2; ++qt) {
            // S^T tiles: 16 keys x 16 queries, f32
            v8f st0 = zero8(), st1 = zero8();
            st0 = wmma_bf16(ka[0][0], qf[qt][0], st0);
            st0 = wmma_bf16(ka[0][1], qf[qt][1], st0);
            st1 = wmma_bf16(ka[1][0], qf[qt][0], st1);
            st1 = wmma_bf16(ka[1][1], qf[qt][1], st1);

            const int qxx = qx0 + qt * 16;
            float p0[8], p1[8];
            float bm = -1e30f;
#pragma unroll
            for (int i = 0; i < 8; ++i) {
                // window mask: qxx <= y <= qxx + 2W as one unsigned compare
                const int ya  = yb + kh8 + i;
                const bool va  = (unsigned)(ya - qxx)      <= (unsigned)(2 * W_);
                const bool vbk = (unsigned)(ya + 16 - qxx) <= (unsigned)(2 * W_);
                const float sa = va  ? st0[i] : -1e30f;
                const float sb = vbk ? st1[i] : -1e30f;
                p0[i] = sa; p1[i] = sb;
                bm = fmaxf(bm, fmaxf(sa, sb));
            }
            bm = fmaxf(bm, __shfl_xor(bm, 16, 32));
            const float mnew = fmaxf(mrun[qt], bm);
            const float sc = __expf(mrun[qt] - mnew);
            float rs = 0.0f;
#pragma unroll
            for (int i = 0; i < 8; ++i) {
                p0[i] = (p0[i] > -1e29f) ? __expf(p0[i] - mnew) : 0.0f;
                p1[i] = (p1[i] > -1e29f) ? __expf(p1[i] - mnew) : 0.0f;
                rs += p0[i] + p1[i];
            }
            rs += __shfl_xor(rs, 16, 32);
            lrun[qt] = lrun[qt] * sc + rs;
            mrun[qt] = mnew;

            // Rescale ctx accumulators: ctx C layout has query row
            // (lane>>4)*8+i -> broadcast sc from score layout (lane = query).
            float scb[8];
#pragma unroll
            for (int i = 0; i < 8; ++i)
                scb[i] = __shfl(sc, kh8 + i, 32);
#pragma unroll
            for (int nt = 0; nt < 4; ++nt)
#pragma unroll
                for (int i = 0; i < 8; ++i)
                    acc[qt][nt][i] *= scb[i];

            // Pack P as bf16 A-fragment (16 queries x 32 keys); the S^T C
            // layout feeds the A halves directly.
            v16bf pa;
#pragma unroll
            for (int j = 0; j < 8; ++j) {
                pa[j]     = (bf16_t)p0[j];
                pa[8 + j] = (bf16_t)p1[j];
            }
#pragma unroll
            for (int nt = 0; nt < 4; ++nt)
                acc[qt][nt] = wmma_bf16(pa, vb[nt], acc[qt][nt]);
        }
    }

    // Epilogue: divide by l (per query row) and store bf16 ctx (b,s,e).
#pragma unroll
    for (int qt = 0; qt < 2; ++qt) {
        float lb[8];
#pragma unroll
        for (int i = 0; i < 8; ++i)
            lb[i] = __shfl(lrun[qt], kh8 + i, 32);
#pragma unroll
        for (int nt = 0; nt < 4; ++nt) {
            const int col = h * D_ + nt * 16 + (lane & 15);
#pragma unroll
            for (int i = 0; i < 8; ++i) {
                const float val = acc[qt][nt][i] / lb[i];
                const int row = c * W_ + qbase + qt * 16 + kh8 + i;
                ctx[(size_t)(b * S_ + row) * E_ + col] = (bf16_t)val;
            }
        }
    }
}

// ---------------------------------------------------------------------------
// Launch
// ---------------------------------------------------------------------------
extern "C" void kernel_launch(void* const* d_in, const int* in_sizes, int n_in,
                              void* d_out, int out_size, void* d_ws, size_t ws_size,
                              hipStream_t stream) {
    (void)in_sizes; (void)n_in; (void)out_size; (void)ws_size;
    const float* x  = (const float*)d_in[0];
    const float* Wq = (const float*)d_in[1];
    const float* bq = (const float*)d_in[2];
    const float* Wk = (const float*)d_in[3];
    const float* bk = (const float*)d_in[4];
    const float* Wv = (const float*)d_in[5];
    const float* bv = (const float*)d_in[6];
    const float* Wo = (const float*)d_in[7];
    const float* bo = (const float*)d_in[8];
    float* out = (float*)d_out;

    const size_t NX = (size_t)B_ * S_ * E_;   // 8,388,608
    const size_t NW = (size_t)E_ * E_;        // 1,048,576

    char* ws = (char*)d_ws;
    size_t off = 0;
    auto take = [&](size_t bytes) -> char* {
        char* p = ws + off;
        off += (bytes + 255) & ~(size_t)255;
        return p;
    };
    bf16_t* xbf  = (bf16_t*)take(NX * 2);
    bf16_t* wqb  = (bf16_t*)take(NW * 2);
    bf16_t* wkb  = (bf16_t*)take(NW * 2);
    bf16_t* wvb  = (bf16_t*)take(NW * 2);
    bf16_t* wob  = (bf16_t*)take(NW * 2);
    bf16_t* qb   = (bf16_t*)take(NX * 2);
    bf16_t* kb   = (bf16_t*)take(NX * 2);
    bf16_t* vtb  = (bf16_t*)take(NX * 2);   // V stored (b, e, s)
    bf16_t* ctxb = (bf16_t*)take(NX * 2);

    const int M = B_ * S_;   // 8192
    const int N = E_;        // 1024
    const int K = E_;        // 1024
    const int gemm_grid = (M / 128) * (N / 128);   // 512

    cvt_f32_bf16<<<(int)((NX + 255) / 256), 256, 0, stream>>>(x,  xbf, (int)NX);
    cvt_f32_bf16<<<(int)((NW + 255) / 256), 256, 0, stream>>>(Wq, wqb, (int)NW);
    cvt_f32_bf16<<<(int)((NW + 255) / 256), 256, 0, stream>>>(Wk, wkb, (int)NW);
    cvt_f32_bf16<<<(int)((NW + 255) / 256), 256, 0, stream>>>(Wv, wvb, (int)NW);
    cvt_f32_bf16<<<(int)((NW + 255) / 256), 256, 0, stream>>>(Wo, wob, (int)NW);

    gemm_wmma<0><<<gemm_grid, 256, 0, stream>>>(xbf, wqb, bq, qb,  M, N, K);
    gemm_wmma<0><<<gemm_grid, 256, 0, stream>>>(xbf, wkb, bk, kb,  M, N, K);
    gemm_wmma<1><<<gemm_grid, 256, 0, stream>>>(xbf, wvb, bv, vtb, M, N, K);

    const int rope_total = B_ * S_ * H_ * (D_ / 2);
    rope_kernel<<<(rope_total + 255) / 256, 256, 0, stream>>>(qb, 0.125f); // 1/sqrt(64)
    rope_kernel<<<(rope_total + 255) / 256, 256, 0, stream>>>(kb, 1.0f);

    attn_kernel<<<B_ * H_ * NC_, 256, 0, stream>>>(qb, kb, vtb, ctxb);

    gemm_wmma<2><<<gemm_grid, 256, 0, stream>>>(ctxb, wob, bo, out, M, N, K);
}
